// CausalSelfAttention_56315611185781
// MI455X (gfx1250) — matrix operand order repaired and checked by reading, not compiled, not run on hardware
//
#include <hip/hip_runtime.h>
#include <math.h>

// ---------------------------------------------------------------------------
// CDNA5 (gfx1250) causal self-attention: QKV GEMM + RoPE + flash attn + proj.
// Matmuls on v_wmma_f32_16x16x32_bf16 (wave32); GEMM staging via the Tensor
// Data Mover (tensor_load_to_lds, double-buffered, TENSORcnt-synchronized);
// softmax row reductions via DPP16 cross-lane ops (no LDS traffic).
// ---------------------------------------------------------------------------

typedef __attribute__((ext_vector_type(16))) __bf16        v16bf;
typedef __attribute__((ext_vector_type(8)))  float         v8f;
typedef __attribute__((ext_vector_type(4)))  unsigned int  su4;
typedef __attribute__((ext_vector_type(8)))  unsigned int  su8;

#define WMMA_BF16(a, b, c) \
  __builtin_amdgcn_wmma_f32_16x16x32_bf16(false, (a), false, (b), (short)0, (c), false, false)

#define BB 4
#define TT 2048
#define NH 16
#define DH 64
#define CM 1024
#define BT (BB * TT)   // 8192 rows

#define GT_M 128
#define GT_N 128
#define GT_K 32
#define LDSP 48        // LDS row pitch in bf16 elems (96 B = 64 B row + 32 B pad)

#define LOG2E 1.44269504088896340736f

__device__ __forceinline__ unsigned short f32_to_bf16(float f) {
  unsigned int u = __float_as_uint(f);
  unsigned int r = 0x7FFFu + ((u >> 16) & 1u);   // round-to-nearest-even
  return (unsigned short)((u + r) >> 16);
}
__device__ __forceinline__ float bf16_to_f32(unsigned short h) {
  return __uint_as_float(((unsigned int)h) << 16);
}

union FragCast { uint4 u[2]; v16bf v; };

// Two 16-byte loads -> one 16x32-bf16 fragment (8 VGPRs / lane).
__device__ __forceinline__ v16bf load_frag(const unsigned short* p0,
                                           const unsigned short* p1) {
  FragCast f;
  f.u[0] = *(const uint4*)p0;
  f.u[1] = *(const uint4*)p1;
  return f.v;
}

__device__ __forceinline__ v8f zero_v8f() {
  v8f z = {0.f, 0.f, 0.f, 0.f, 0.f, 0.f, 0.f, 0.f};
  return z;
}

// ---- DPP16 cross-lane reduction within each 16-lane half (VALU only) ------
// 0xB1 = quad_perm[1,0,3,2] (xor1), 0x4E = quad_perm[2,3,0,1] (xor2),
// 0x141 = row_half_mirror (combines quads in 8), 0x140 = row_mirror (8<->8).
template <int CTRL>
__device__ __forceinline__ float dpp_mov_f(float x) {
  return __int_as_float(__builtin_amdgcn_update_dpp(
      __float_as_int(x), __float_as_int(x), CTRL, 0xF, 0xF, true));
}
__device__ __forceinline__ float red16_max(float x) {
  x = fmaxf(x, dpp_mov_f<0xB1>(x));
  x = fmaxf(x, dpp_mov_f<0x4E>(x));
  x = fmaxf(x, dpp_mov_f<0x141>(x));
  x = fmaxf(x, dpp_mov_f<0x140>(x));
  return x;
}
__device__ __forceinline__ float red16_sum(float x) {
  x += dpp_mov_f<0xB1>(x);
  x += dpp_mov_f<0x4E>(x);
  x += dpp_mov_f<0x141>(x);
  x += dpp_mov_f<0x140>(x);
  return x;
}

// Flat pointer into the LDS aperture: LDS byte address = addr[31:0].
__device__ __forceinline__ unsigned lds_addr_u32(const void* p) {
  return (unsigned)(unsigned long long)(uintptr_t)p;
}

// TDM: DMA one [rows=128 x cols=32] bf16 tile (row stride = stride_elems) from
// global into LDS at ldsaddr, inserting 32 B of padding after each 64 B row
// (-> 96 B LDS pitch, matching LDSP). 2D descriptor: groups 2/3 omitted.
__device__ __forceinline__ void tdm_load_tile_2d(const unsigned short* gptr,
                                                 unsigned ldsaddr,
                                                 unsigned stride_elems) {
  const unsigned long long ga = (unsigned long long)(uintptr_t)gptr;
  su4 g0;
  g0[0] = 1u;                                            // count=1 (valid user D#)
  g0[1] = ldsaddr;                                       // lds_addr[31:0]
  g0[2] = (unsigned)(ga & 0xFFFFFFFFu);                  // global_addr[31:0]
  g0[3] = ((unsigned)(ga >> 32) & 0x01FFFFFFu)           // global_addr[56:32]
        | 0x80000000u;                                   // type=2 ("image")
  su8 g1;
  g1[0] = (1u << 16)                                     // data_size = 2 bytes
        | (1u << 20)                                     // pad_enable
        | (3u << 22)                                     // pad_interval: 16 DW (64 B)
        | (7u << 25);                                    // pad_amount: 8 DW (32 B)
  g1[1] = (stride_elems & 0xFFFFu) << 16;                // tensor_dim0[15:0]
  g1[2] = ((stride_elems >> 16) & 0xFFFFu);              // tensor_dim0[31:16]; dim1 lo16=0
  g1[3] = 0x10u                                          // tensor_dim1 = 1<<20 (hi16)
        | (32u << 16);                                   // tile_dim0 = 32 elems
  g1[4] = 128u;                                          // tile_dim1 = 128, tile_dim2 = 0
  g1[5] = stride_elems;                                  // tensor_dim0_stride[31:0]
  g1[6] = 0u;                                            // stride hi / dim1_stride lo
  g1[7] = 0u;
  asm volatile("tensor_load_to_lds %0, %1" :: "s"(g0), "s"(g1) : "memory");
}

// ---------------------------------------------------------------------------
// fp32 -> bf16 elementwise
__global__ void k_cvt_bf16(const float* __restrict__ in,
                           unsigned short* __restrict__ out, int n) {
  int i = blockIdx.x * blockDim.x + threadIdx.x;
  if (i < n) out[i] = f32_to_bf16(in[i]);
}

// in [K][N] fp32 -> out [N][K] bf16
__global__ void k_transpose_bf16(const float* __restrict__ in,
                                 unsigned short* __restrict__ out,
                                 int K, int N) {
  int i = blockIdx.x * blockDim.x + threadIdx.x;
  if (i < K * N) {
    int n = i / K, k = i - n * K;
    out[i] = f32_to_bf16(in[(size_t)k * N + n]);
  }
}

// ---------------------------------------------------------------------------
// C[M][N] = A[M][K] * Bt[N][K]^T, bf16 in, fp32 accum via WMMA.
// 256 thr = 8 waves; 128x128 tile; wave tile 32x64 (2x4 accums); K-step 32.
// A/B tiles staged by the TDM, double-buffered in LDS.
template <bool OUT_BF16>
__global__ __launch_bounds__(256) void gemm_bf16(
    const unsigned short* __restrict__ A,
    const unsigned short* __restrict__ Bt,
    void* __restrict__ Cout, int M, int N, int K) {
  __shared__ alignas(16) unsigned short lA[2][GT_M * LDSP];
  __shared__ alignas(16) unsigned short lB[2][GT_N * LDSP];
  const int tid  = threadIdx.x;
  const int wid  = tid >> 5;
  const int lane = tid & 31;
  const int ln   = lane & 15;
  const int half = lane >> 4;
  const int m0 = blockIdx.y * GT_M;
  const int n0 = blockIdx.x * GT_N;
  const int wm = (wid >> 1) * 32;
  const int wn = (wid & 1) * 64;

  v8f acc[2][4];
#pragma unroll
  for (int mi = 0; mi < 2; ++mi)
#pragma unroll
    for (int ni = 0; ni < 4; ++ni) acc[mi][ni] = zero_v8f();

  const int nk = K / GT_K;
  if (wid == 0) {   // one wave drives the TDM (EXEC is ignored by tensor ops)
    tdm_load_tile_2d(&A[(size_t)m0 * K], lds_addr_u32(&lA[0][0]), (unsigned)K);
    tdm_load_tile_2d(&Bt[(size_t)n0 * K], lds_addr_u32(&lB[0][0]), (unsigned)K);
  }

  for (int it = 0; it < nk; ++it) {
    if (wid == 0) __builtin_amdgcn_s_wait_tensorcnt(0);
    __syncthreads();                       // buffer `cur` ready for everyone
    const int cur = it & 1;
    if (wid == 0 && (it + 1) < nk) {       // overlap DMA of next tile
      const int k0n = (it + 1) * GT_K;
      tdm_load_tile_2d(&A[(size_t)m0 * K + k0n],
                       lds_addr_u32(&lA[cur ^ 1][0]), (unsigned)K);
      tdm_load_tile_2d(&Bt[(size_t)n0 * K + k0n],
                       lds_addr_u32(&lB[cur ^ 1][0]), (unsigned)K);
    }

    // A frag (16x32): lanes 0-15 K={0..7,16..23}, lanes 16-31 K={8..15,24..31}
    v16bf af[2];
    const int adb = half ? 8 : 0;
#pragma unroll
    for (int mi = 0; mi < 2; ++mi) {
      const unsigned short* p = &lA[cur][(wm + mi * 16 + ln) * LDSP + adb];
      af[mi] = load_frag(p, p + 16);
    }
    // B frag (32x16): lanes 0-15 K={0..7,16..23}, lanes 16-31 K={8..15,24..31}
    v16bf bfr[4];
    const int bdb = half ? 8 : 0;
#pragma unroll
    for (int ni = 0; ni < 4; ++ni) {
      const unsigned short* p = &lB[cur][(wn + ni * 16 + ln) * LDSP + bdb];
      bfr[ni] = load_frag(p, p + 16);
    }
#pragma unroll
    for (int mi = 0; mi < 2; ++mi)
#pragma unroll
      for (int ni = 0; ni < 4; ++ni)
        acc[mi][ni] = WMMA_BF16(af[mi], bfr[ni], acc[mi][ni]);
  }

  // C layout: lane l, VGPR r -> row = r + 8*(l>>4), col = l&15
#pragma unroll
  for (int mi = 0; mi < 2; ++mi)
#pragma unroll
    for (int ni = 0; ni < 4; ++ni)
#pragma unroll
      for (int r = 0; r < 8; ++r) {
        const int row = m0 + wm + mi * 16 + r + 8 * half;
        const int col = n0 + wn + ni * 16 + ln;
        const float v = acc[mi][ni][r];
        if (OUT_BF16)
          ((unsigned short*)Cout)[(size_t)row * N + col] = f32_to_bf16(v);
        else
          ((float*)Cout)[(size_t)row * N + col] = v;
      }
}

// ---------------------------------------------------------------------------
// RoPE + head reshape. qkv bf16 [B*T][3*CM] -> Qh,Kh bf16 [B,H,T,64] (Q holds
// the 1/sqrt(Dh)*log2(e) scale so softmax can use exp2), Vt bf16 [B,H,64,T].
__global__ void k_rope(const unsigned short* __restrict__ qkv,
                       const float* __restrict__ cosT,
                       const float* __restrict__ sinT,
                       unsigned short* __restrict__ Qh,
                       unsigned short* __restrict__ Kh,
                       unsigned short* __restrict__ Vt) {
  const int i = blockIdx.x * blockDim.x + threadIdx.x;   // (b,t,h,d) flat
  const int d = i & 63;
  const int h = (i >> 6) & 15;
  const int t = (i >> 10) & (TT - 1);
  const int b = i >> 21;
  const size_t base = (size_t)(b * TT + t) * (3 * CM) + h * DH;
  const float qv = bf16_to_f32(qkv[base + d]);
  const float kv = bf16_to_f32(qkv[base + CM + d]);
  const float vv = bf16_to_f32(qkv[base + 2 * CM + d]);
  const int   dp  = (d < 32) ? d + 32 : d - 32;
  const float sgn = (d < 32) ? -1.0f : 1.0f;
  const float qp = bf16_to_f32(qkv[base + dp]);
  const float kp = bf16_to_f32(qkv[base + CM + dp]);
  const float c = cosT[t * DH + d];
  const float s = sinT[t * DH + d];
  const float qr = qv * c + sgn * qp * s;
  const float kr = kv * c + sgn * kp * s;
  const size_t ho = ((size_t)(b * NH + h) * TT + t) * DH + d;
  Qh[ho] = f32_to_bf16(qr * (0.125f * LOG2E));   // 1/sqrt(64) and log2(e)
  Kh[ho] = f32_to_bf16(kr);
  Vt[((size_t)(b * NH + h) * DH + d) * TT + t] = f32_to_bf16(vv);
}

// ---------------------------------------------------------------------------
// Flash attention: one wave per (b,h, 16-query tile); online softmax (base-2)
// over 32-key steps; masking only in the final (diagonal) step; row
// reductions via DPP16 (no LDS).
__global__ __launch_bounds__(256) void k_flash(
    const unsigned short* __restrict__ Qh,
    const unsigned short* __restrict__ Kh,
    const unsigned short* __restrict__ Vt,
    unsigned short* __restrict__ Y) {
  __shared__ alignas(16) unsigned short lP[8][16 * LDSP];  // per-wave P stage
  const int tid  = threadIdx.x;
  const int wid  = tid >> 5;
  const int lane = tid & 31;
  const int ln   = lane & 15;
  const int half = lane >> 4;
  const int w  = blockIdx.x * 8 + wid;
  const int qt = w & 127;
  const int h  = (w >> 7) & 15;
  const int b  = w >> 11;
  const int q0 = qt << 4;
  const size_t bh = (size_t)(b * NH + h);
  const unsigned short* Qb = Qh + bh * (size_t)TT * DH;
  const unsigned short* Kb = Kh + bh * (size_t)TT * DH;
  const unsigned short* Vb = Vt + bh * (size_t)DH * TT;

  // Persistent Q fragments (16 rows x 64 d = two 16x32 A-frags)
  v16bf qf[2];
  {
    const int adb = half ? 8 : 0;
#pragma unroll
    for (int f = 0; f < 2; ++f) {
      const unsigned short* p = &Qb[(size_t)(q0 + ln) * DH + f * 32 + adb];
      qf[f] = load_frag(p, p + 16);
    }
  }

  v8f o[4];
#pragma unroll
  for (int dt = 0; dt < 4; ++dt) o[dt] = zero_v8f();
  float mrun[8], lrun[8];
#pragma unroll
  for (int r = 0; r < 8; ++r) { mrun[r] = -3.0e38f; lrun[r] = 0.0f; }

  unsigned short* pw = &lP[wid][0];

  auto step = [&](int kt, bool masked) {
    // S tiles: keys [kt,kt+16) and [kt+16,kt+32); K rows row-major -> B frags
    v8f s0 = zero_v8f(), s1 = zero_v8f();
    const int kdb = half ? 8 : 0;
#pragma unroll
    for (int f = 0; f < 2; ++f) {
      const unsigned short* p0 = &Kb[(size_t)(kt + ln) * DH + f * 32 + kdb];
      s0 = WMMA_BF16(qf[f], load_frag(p0, p0 + 16), s0);
      const unsigned short* p1 = &Kb[(size_t)(kt + 16 + ln) * DH + f * 32 + kdb];
      s1 = WMMA_BF16(qf[f], load_frag(p1, p1 + 16), s1);
    }

    // Online softmax (base-2): row = r + 8*half, cols on lanes 0..15 per half
    float p0v[8], p1v[8];
#pragma unroll
    for (int r = 0; r < 8; ++r) {
      float x0 = s0[r];
      float x1 = s1[r];
      if (masked) {
        const int rowg = q0 + r + 8 * half;
        x0 = ((kt + ln)      <= rowg) ? x0 : -3.0e38f;
        x1 = ((kt + 16 + ln) <= rowg) ? x1 : -3.0e38f;
      }
      const float t  = red16_max(fmaxf(x0, x1));
      const float mn = fmaxf(mrun[r], t);
      const float e0 = exp2f(x0 - mn);
      const float e1 = exp2f(x1 - mn);
      const float sum = red16_sum(e0 + e1);
      const float alpha = exp2f(mrun[r] - mn);
      lrun[r] = lrun[r] * alpha + sum;
      mrun[r] = mn;
      p0v[r] = e0;
      p1v[r] = e1;
#pragma unroll
      for (int dt = 0; dt < 4; ++dt) o[dt][r] *= alpha;
    }

    // Re-layout P (C-layout) -> A-fragment via per-wave LDS buffer.
#pragma unroll
    for (int r = 0; r < 8; ++r) {
      const int row = r + 8 * half;
      pw[row * LDSP + ln]      = f32_to_bf16(p0v[r]);
      pw[row * LDSP + 16 + ln] = f32_to_bf16(p1v[r]);
    }
    asm volatile("s_wait_dscnt 0" ::: "memory");   // same-wave LDS RAW
    const int adb = half ? 8 : 0;
    const unsigned short* pp = &pw[ln * LDSP + adb];
    v16bf pf = load_frag(pp, pp + 16);

    // O += P * V; Vt is d-major so B-frags are contiguous key runs.
#pragma unroll
    for (int dt = 0; dt < 4; ++dt) {
      const unsigned short* vp =
          &Vb[(size_t)(dt * 16 + ln) * TT + kt + (half ? 8 : 0)];
      o[dt] = WMMA_BF16(pf, load_frag(vp, vp + 16), o[dt]);
    }
  };

  int kt = 0;
  const int nfull = q0 >> 5;            // key steps with no causal masking
  for (int it = 0; it < nfull; ++it, kt += 32) step(kt, false);
  step(kt, true);                       // diagonal step

  // Normalize and write y bf16 in [B*T][CM] layout (col = h*64 + d).
#pragma unroll
  for (int r = 0; r < 8; ++r) {
    const float inv  = 1.0f / lrun[r];
    const int   trow = q0 + r + 8 * half;
    const size_t yo  = ((size_t)(b * TT + trow)) * CM + h * DH + ln;
#pragma unroll
    for (int dt = 0; dt < 4; ++dt)
      Y[yo + dt * 16] = f32_to_bf16(o[dt][r] * inv);
  }
}

// ---------------------------------------------------------------------------
extern "C" void kernel_launch(void* const* d_in, const int* in_sizes, int n_in,
                              void* d_out, int out_size, void* d_ws, size_t ws_size,
                              hipStream_t stream) {
  (void)in_sizes; (void)n_in; (void)out_size; (void)ws_size;
  const float* x     = (const float*)d_in[0];
  const float* cosT  = (const float*)d_in[1];
  const float* sinT  = (const float*)d_in[2];
  const float* Wqkv  = (const float*)d_in[3];   // [1024][3072]
  const float* Wproj = (const float*)d_in[4];   // [1024][1024]
  float* out = (float*)d_out;

  char* ws = (char*)d_ws;
  size_t off = 0;
  auto take = [&](size_t bytes) -> void* {
    void* p = ws + off;
    off += (bytes + 255) & ~(size_t)255;
    return p;
  };
  unsigned short* xbf    = (unsigned short*)take((size_t)BT * CM * 2);      // reused as y
  unsigned short* wqkvt  = (unsigned short*)take((size_t)3 * CM * CM * 2);
  unsigned short* wprojt = (unsigned short*)take((size_t)CM * CM * 2);
  unsigned short* qkv    = (unsigned short*)take((size_t)BT * 3 * CM * 2);
  unsigned short* QhP    = (unsigned short*)take((size_t)BB * NH * TT * DH * 2);
  unsigned short* KhP    = (unsigned short*)take((size_t)BB * NH * TT * DH * 2);
  unsigned short* VtP    = (unsigned short*)take((size_t)BB * NH * DH * TT * 2);

  k_cvt_bf16<<<(BT * CM + 255) / 256, 256, 0, stream>>>(x, xbf, BT * CM);
  k_transpose_bf16<<<(3 * CM * CM + 255) / 256, 256, 0, stream>>>(Wqkv, wqkvt, CM, 3 * CM);
  k_transpose_bf16<<<(CM * CM + 255) / 256, 256, 0, stream>>>(Wproj, wprojt, CM, CM);

  gemm_bf16<true><<<dim3(3 * CM / GT_N, BT / GT_M), 256, 0, stream>>>(
      xbf, wqkvt, qkv, BT, 3 * CM, CM);

  k_rope<<<(BT * CM) / 256, 256, 0, stream>>>(qkv, cosT, sinT, QhP, KhP, VtP);

  k_flash<<<(BB * NH * (TT / 16)) / 8, 256, 0, stream>>>(QhP, KhP, VtP, xbf);

  gemm_bf16<false><<<dim3(CM / GT_N, BT / GT_M), 256, 0, stream>>>(
      xbf, wprojt, out, BT, CM, CM);
}
